// KernelNN_52896817218081
// MI455X (gfx1250) — matrix operand order repaired and checked by reading, compiled, not run on hardware
//
#include <hip/hip_runtime.h>

#define N_NODES 961
#define N_EDGES 32768
#define WIDTH   64
#define KER_W   1024
#define KER_IN  6
#define DEPTH   6

typedef __bf16 bf16;
typedef __attribute__((ext_vector_type(16))) __bf16 v16bf;
typedef __attribute__((ext_vector_type(8)))  __bf16 v8bf;
typedef __attribute__((ext_vector_type(8)))  float  v8f;

__device__ __forceinline__ bf16 f2bf(float f) {
    union { float f; unsigned u; } in; in.f = f;
    unsigned r = in.u + 0x7FFFu + ((in.u >> 16) & 1u);   // round-to-nearest-even
    union { unsigned short s; bf16 b; } out; out.s = (unsigned short)(r >> 16);
    return out.b;
}
__device__ __forceinline__ float bfbits_lo(unsigned p) {
    union { unsigned u; float f; } c; c.u = p << 16; return c.f; }
__device__ __forceinline__ float bfbits_hi(unsigned p) {
    union { unsigned u; float f; } c; c.u = p & 0xFFFF0000u; return c.f; }

// ---------------------------------------------------------------- utilities
__global__ void k_zero(float* __restrict__ p, int n) {
    int t = blockIdx.x * blockDim.x + threadIdx.x;
    if (t < n) p[t] = 0.f;
}

// Wt[n][k] = bf16(W[k][n]);  W is [1024][N] fp32, Wt is [N][1024] bf16
__global__ void k_conv_wT(const float* __restrict__ W, bf16* __restrict__ Wt, int N) {
    int t = blockIdx.x * blockDim.x + threadIdx.x;
    int n = t >> 10, k = t & 1023;
    if (n < N) Wt[(size_t)n * 1024 + k] = f2bf(W[(size_t)k * N + n]);
}

// ---------------------------------------------------------------- layer 1 (K=6, tiny)
__global__ void k_layer1(const float* __restrict__ ea, const float* __restrict__ w1,
                         const float* __restrict__ b1, bf16* __restrict__ h1) {
    __shared__ float a[KER_IN];
    int e = blockIdx.x;
    if (threadIdx.x < KER_IN) a[threadIdx.x] = ea[(size_t)e * KER_IN + threadIdx.x];
    __syncthreads();
#pragma unroll
    for (int r = 0; r < 4; ++r) {
        int j = threadIdx.x + r * 256;
        float s = b1[j];
#pragma unroll
        for (int k = 0; k < KER_IN; ++k) s += a[k] * w1[k * KER_W + j];
        h1[(size_t)e * KER_W + j] = f2bf(s > 0.f ? s : 0.f);
    }
}

// ---------------------------------------------------------------- bf16 WMMA GEMM
// C[M,N] = relu?(A[M,K] @ Bt[N,K]^T + bias); fp32 accumulation.
// Block: 256 threads = 8 waves. Wave tile: 32(M) x 64(N) (2 A frags x 4 B frags,
// 8 WMMAs per k-step). Block tile: 256(M) x 64(N).
// Software-pipelined: fragment double-buffering so next k-step loads are in
// flight while current WMMAs execute (partial loadcnt waits instead of 0).
// NOTE: the final prefetch reads 32 elements past K; all operand buffers are
// followed by other live workspace allocations, so the overrun stays in d_ws.
__global__ __launch_bounds__(256)
void k_gemm_bf16(const bf16* __restrict__ A, const bf16* __restrict__ Bt,
                 const float* __restrict__ bias, bf16* __restrict__ C,
                 int M, int N, int K, int relu) {
    const int wave = threadIdx.x >> 5;
    const int lane = threadIdx.x & 31;
    const int lh   = lane >> 4;           // half select (0: lanes 0-15, 1: 16-31)
    const int ll   = lane & 15;
    const int m0   = blockIdx.y * 256 + wave * 32;
    const int n0   = blockIdx.x * 64;

    // A frag (16x32, row = m + ll): lanes 0-15 K {0..7,16..23}, lanes 16-31 K {8..15,24..31}
    auto loadA = [&](int ms, int k) {
        const bf16* p = A + (size_t)(m0 + ms * 16 + ll) * K + lh * 8 + k;
        v8bf lo = *(const v8bf*)(p);
        v8bf hi = *(const v8bf*)(p + 16);
        v16bf r;
#pragma unroll
        for (int i = 0; i < 8; ++i) { r[i] = lo[i]; r[i + 8] = hi[i]; }
        return r;
    };
    // B frag (32x16, col = n + ll): lanes 0-15 K {0..7,16..23}, lanes 16-31 K {8..15,24..31}
    auto loadB = [&](int t, int k) {
        const bf16* p = Bt + (size_t)(n0 + t * 16 + ll) * K + lh * 8 + k;
        v8bf lo = *(const v8bf*)(p);
        v8bf hi = *(const v8bf*)(p + 16);
        v16bf r;
#pragma unroll
        for (int i = 0; i < 8; ++i) { r[i] = lo[i]; r[i + 8] = hi[i]; }
        return r;
    };

    const v8f zc = {0.f, 0.f, 0.f, 0.f, 0.f, 0.f, 0.f, 0.f};
    v8f c[2][4] = {{zc, zc, zc, zc}, {zc, zc, zc, zc}};

    v16bf A0[2], B0[4], A1[2], B1[4];
    A0[0] = loadA(0, 0); A0[1] = loadA(1, 0);
#pragma unroll
    for (int t = 0; t < 4; ++t) B0[t] = loadB(t, 0);

    const int steps = K / 64;             // K = 1024 -> 16 double-steps
    for (int i = 0; i < steps; ++i) {
        const int k1 = i * 64 + 32;
        const int k2 = i * 64 + 64;       // == K on last iter (safe overrun, see note)
        A1[0] = loadA(0, k1); A1[1] = loadA(1, k1);
#pragma unroll
        for (int t = 0; t < 4; ++t) B1[t] = loadB(t, k1);
#pragma unroll
        for (int ms = 0; ms < 2; ++ms)
#pragma unroll
            for (int t = 0; t < 4; ++t)
                c[ms][t] = __builtin_amdgcn_wmma_f32_16x16x32_bf16(
                    false, A0[ms], false, B0[t], (short)0, c[ms][t], false, false);
        A0[0] = loadA(0, k2); A0[1] = loadA(1, k2);
#pragma unroll
        for (int t = 0; t < 4; ++t) B0[t] = loadB(t, k2);
#pragma unroll
        for (int ms = 0; ms < 2; ++ms)
#pragma unroll
            for (int t = 0; t < 4; ++t)
                c[ms][t] = __builtin_amdgcn_wmma_f32_16x16x32_bf16(
                    false, A1[ms], false, B1[t], (short)0, c[ms][t], false, false);
    }

#pragma unroll
    for (int ms = 0; ms < 2; ++ms)
#pragma unroll
        for (int t = 0; t < 4; ++t) {
            const int col = n0 + t * 16 + ll;
            const float bv = bias[col];
#pragma unroll
            for (int j = 0; j < 8; ++j) {   // C layout: VGPR j -> M = j (+8 for hi lanes)
                float v = c[ms][t][j] + bv;
                if (relu) v = v > 0.f ? v : 0.f;
                C[(size_t)(m0 + ms * 16 + j + lh * 8) * N + col] = f2bf(v);
            }
        }
}

// ---------------------------------------------------------------- node-side kernels
__global__ void k_fc1(const float* __restrict__ x, const float* __restrict__ w,
                      const float* __restrict__ b, float* __restrict__ z) {
    int t = blockIdx.x * blockDim.x + threadIdx.x;
    if (t < N_NODES * WIDTH) {
        int n = t >> 6, j = t & 63;
        z[t] = x[n] * w[j] + b[j];
    }
}

__global__ void k_count(const int* __restrict__ ei, float* __restrict__ cnt) {
    int e = blockIdx.x * blockDim.x + threadIdx.x;
    if (e < N_EDGES) atomicAdd(&cnt[ei[N_EDGES + e]], 1.f);
}

// One wave per edge: msg = z[src] (1x64) @ W_e (64x64 bf16) -> atomic scatter to agg[dst]
__global__ __launch_bounds__(256)
void k_message(const float* __restrict__ z, const bf16* __restrict__ Wm,
               const int* __restrict__ ei, float* __restrict__ agg) {
    __shared__ float zsh[8][WIDTH];
    const int wave = threadIdx.x >> 5, lane = threadIdx.x & 31;
    const int e = blockIdx.x * 8 + wave;
    const int src = ei[e], dst = ei[N_EDGES + e];
    const float2 zp = *(const float2*)(z + (size_t)src * WIDTH + lane * 2);
    zsh[wave][lane * 2] = zp.x; zsh[wave][lane * 2 + 1] = zp.y;
    __syncthreads();
    const unsigned* wrow = (const unsigned*)(Wm + (size_t)e * (WIDTH * WIDTH));
    float a0 = 0.f, a1 = 0.f;
#pragma unroll 8
    for (int i = 0; i < WIDTH; ++i) {     // lane handles out cols {2*lane, 2*lane+1}
        const float zi = zsh[wave][i];
        const unsigned p = wrow[i * 32 + lane];
        a0 += zi * bfbits_lo(p);
        a1 += zi * bfbits_hi(p);
    }
    atomicAdd(&agg[(size_t)dst * WIDTH + lane * 2], a0);
    atomicAdd(&agg[(size_t)dst * WIDTH + lane * 2 + 1], a1);
}

// z = relu(agg/cnt + z@root + conv_b); one wave per node
__global__ __launch_bounds__(256)
void k_update(const float* __restrict__ agg, const float* __restrict__ cnt,
              const float* __restrict__ zin, const float* __restrict__ root,
              const float* __restrict__ cb, float* __restrict__ zout) {
    __shared__ float zsh[8][WIDTH];
    const int wave = threadIdx.x >> 5, lane = threadIdx.x & 31;
    const int n = blockIdx.x * 8 + wave;
    const bool ok = n < N_NODES;
    if (ok) {
        const float2 zp = *(const float2*)(zin + (size_t)n * WIDTH + lane * 2);
        zsh[wave][lane * 2] = zp.x; zsh[wave][lane * 2 + 1] = zp.y;
    }
    __syncthreads();
    if (!ok) return;
    const float inv = 1.f / fmaxf(cnt[n], 1.f);
    const int o0 = lane * 2;
    float r0 = agg[(size_t)n * WIDTH + o0] * inv + cb[o0];
    float r1 = agg[(size_t)n * WIDTH + o0 + 1] * inv + cb[o0 + 1];
#pragma unroll 8
    for (int i = 0; i < WIDTH; ++i) {
        const float zi = zsh[wave][i];
        const float2 rp = *(const float2*)(root + i * WIDTH + o0);
        r0 += zi * rp.x; r1 += zi * rp.y;
    }
    zout[(size_t)n * WIDTH + o0]     = fmaxf(r0, 0.f);
    zout[(size_t)n * WIDTH + o0 + 1] = fmaxf(r1, 0.f);
}

__global__ void k_fc2(const float* __restrict__ z, const float* __restrict__ w,
                      const float* __restrict__ b, float* __restrict__ out) {
    int n = blockIdx.x * blockDim.x + threadIdx.x;
    if (n < N_NODES) {
        float s = b[0];
#pragma unroll 8
        for (int j = 0; j < WIDTH; ++j) s += z[(size_t)n * WIDTH + j] * w[j];
        out[n] = s;
    }
}

// ---------------------------------------------------------------- driver
extern "C" void kernel_launch(void* const* d_in, const int* in_sizes, int n_in,
                              void* d_out, int out_size, void* d_ws, size_t ws_size,
                              hipStream_t stream) {
    const float* x     = (const float*)d_in[0];
    const int*   ei    = (const int*)d_in[1];
    const float* ea    = (const float*)d_in[2];
    const float* fc1_w = (const float*)d_in[3];
    const float* fc1_b = (const float*)d_in[4];
    const float* k_w1  = (const float*)d_in[5];
    const float* k_b1  = (const float*)d_in[6];
    const float* k_w2  = (const float*)d_in[7];
    const float* k_b2  = (const float*)d_in[8];
    const float* k_w3  = (const float*)d_in[9];
    const float* k_b3  = (const float*)d_in[10];
    const float* root  = (const float*)d_in[11];
    const float* convb = (const float*)d_in[12];
    const float* fc2_w = (const float*)d_in[13];
    const float* fc2_b = (const float*)d_in[14];
    float* out = (float*)d_out;

    // Workspace layout (~395 MB total). Order matters: GEMM prefetch may read
    // up to 64 B past each operand buffer; every buffer below is followed by
    // another live allocation, so overruns stay inside d_ws.
    char* ws = (char*)d_ws;
    bf16* h1   = (bf16*)(ws);                                    //  64 MB  [E,1024] bf16
    bf16* h2   = (bf16*)(ws + (64ull  << 20));                   //  64 MB  [E,1024] bf16
    bf16* Wm   = (bf16*)(ws + (128ull << 20));                   // 256 MB  [E,64,64] bf16
    bf16* Wt2  = (bf16*)(ws + (384ull << 20));                   //   2 MB  [1024,1024]^T
    bf16* Wt3  = (bf16*)(ws + (386ull << 20));                   //   8 MB  [4096,1024]^T
    float* z0  = (float*)(ws + (394ull << 20));                  // [961,64]
    float* z1  = (float*)(ws + (394ull << 20) + (256u << 10));
    float* agg = (float*)(ws + (394ull << 20) + (512u << 10));
    float* cnt = (float*)(ws + (394ull << 20) + (768u << 10));

    // Weights -> transposed bf16 (once per launch; L2-resident afterwards)
    k_conv_wT<<<(1024 * 1024) / 256, 256, 0, stream>>>(k_w2, Wt2, 1024);
    k_conv_wT<<<(4096 * 1024) / 256, 256, 0, stream>>>(k_w3, Wt3, 4096);

    // Kernel MLP: the 344-GFLOP part, on the WMMA path
    k_layer1<<<N_EDGES, 256, 0, stream>>>(ea, k_w1, k_b1, h1);
    k_gemm_bf16<<<dim3(1024 / 64, N_EDGES / 256), 256, 0, stream>>>(
        h1, Wt2, k_b2, h2, N_EDGES, 1024, 1024, 1);
    k_gemm_bf16<<<dim3(4096 / 64, N_EDGES / 256), 256, 0, stream>>>(
        h2, Wt3, k_b3, Wm, N_EDGES, 4096, 1024, 0);

    // Node init + scatter-mean denominators
    k_fc1<<<(N_NODES * WIDTH + 255) / 256, 256, 0, stream>>>(x, fc1_w, fc1_b, z0);
    k_zero<<<(N_NODES + 255) / 256, 256, 0, stream>>>(cnt, N_NODES);
    k_count<<<N_EDGES / 256, 256, 0, stream>>>(ei, cnt);

    // Depth loop: bandwidth-bound per-edge GEMV over materialized bf16 Wmat
    float* zc = z0; float* zn = z1;
    for (int d = 0; d < DEPTH; ++d) {
        k_zero<<<(N_NODES * WIDTH + 255) / 256, 256, 0, stream>>>(agg, N_NODES * WIDTH);
        k_message<<<N_EDGES / 8, 256, 0, stream>>>(zc, Wm, ei, agg);
        k_update<<<(N_NODES + 7) / 8, 256, 0, stream>>>(agg, cnt, zc, root, convb, zn);
        float* t = zc; zc = zn; zn = t;
    }
    k_fc2<<<(N_NODES + 255) / 256, 256, 0, stream>>>(zc, fc2_w, fc2_b, out);
}
